// CampaignSimilarityDetector_74929999446925
// MI455X (gfx1250) — matrix operand order repaired and checked by reading, not compiled, not run on hardware
//
#include <hip/hip_runtime.h>
#include <hip/hip_bf16.h>
#include <math.h>

typedef _Float16 v16h __attribute__((ext_vector_type(16)));
typedef _Float16 v8h  __attribute__((ext_vector_type(8)));
typedef float    v8f  __attribute__((ext_vector_type(8)));

#define NROWS 8192
#define DDIM  256
#define THR   0.85f
#define WPR   (NROWS / 32)          /* adjacency words per row = 256 */

/* keep epilogue mask live-ranges short: stop the scheduler from hoisting
   all v_cmp masks to the top (SGPR pressure -> spills) */
#if __has_builtin(__builtin_amdgcn_sched_barrier)
#define SCHED_FENCE() __builtin_amdgcn_sched_barrier(0)
#else
#define SCHED_FENCE()
#endif

/* workspace layout (bytes) */
#define OFF_NORMED ((size_t)0)                                 /* 8192*256*2 = 4 MiB  */
#define OFF_ADJ    ((size_t)(NROWS * DDIM * 2))                /* 8192*256*4 = 8 MiB  */
#define OFF_LAB0   (OFF_ADJ + (size_t)NROWS * WPR * 4)
#define OFF_LAB1   (OFF_LAB0 + (size_t)NROWS * 4)
#define OFF_STATS  (OFF_LAB1 + (size_t)NROWS * 4)              /* u32 maxkey, u32 cnt, u64 sumfix */

__device__ __forceinline__ unsigned int fkey(float f) {
    unsigned int u = __float_as_uint(f);
    return (u & 0x80000000u) ? ~u : (u | 0x80000000u);
}
__device__ __forceinline__ float fdec(unsigned int k) {
    unsigned int u = (k & 0x80000000u) ? (k & 0x7fffffffu) : ~k;
    return __uint_as_float(u);
}

/* ---------------- init: labels = iota, stats = 0 ---------------- */
__global__ void k_init(int* __restrict__ lab0, unsigned int* __restrict__ su32,
                       unsigned long long* __restrict__ su64) {
    int i = blockIdx.x * blockDim.x + threadIdx.x;
    if (i < NROWS) lab0[i] = i;
    if (i == 0) { su32[0] = 0u; su32[1] = 0u; su64[1] = 0ull; }
}

/* ---------------- row-normalize fp32 -> f16 ---------------- */
__global__ __launch_bounds__(256) void k_normalize(const float* __restrict__ emb,
                                                   _Float16* __restrict__ normed) {
    __shared__ float red[256];
    int row = blockIdx.x, t = threadIdx.x;
    float x = emb[row * DDIM + t];
    red[t] = x * x;
    __syncthreads();
    for (int s = 128; s > 0; s >>= 1) {
        if (t < s) red[t] += red[t + s];
        __syncthreads();
    }
    float nrm = fmaxf(sqrtf(red[0]), 1e-12f);
    normed[row * DDIM + t] = (_Float16)(x / nrm);
}

/* Straight-line tile epilogue, specialized on whether the tile touches the
   diagonal. C layout: VGPR e -> row M=e (lanes 0-15) / M=e+8 (lanes 16-31),
   col=lane&15. Adjacency words are wave-uniform (ballot results); deposit
   them with lane-selects so lane L holds the 2 words of row rowBase0+L. */
template <bool DIAG>
__device__ __forceinline__ void tile_epilogue(const v8f (&acc)[2][4],
                                              int lane, int lm, int hi,
                                              int rowBase0, int colBase0,
                                              float& mx, float& sum, unsigned int& cnt,
                                              unsigned int& wlo, unsigned int& whi) {
#pragma unroll
    for (int smi = 0; smi < 2; smi++) {
#pragma unroll
        for (int e = 0; e < 8; e++) {
            const int rr1 = smi * 16 + e;      /* relative row owned by lanes 0-15  */
            const int rr2 = rr1 + 8;           /* relative row owned by lanes 16-31 */
            unsigned int bal[4];
#pragma unroll
            for (int sn = 0; sn < 4; sn++) {
                const float s  = acc[smi][sn][e];
                const bool hot = (s > THR);
                if (DIAG) {
                    const int row  = rowBase0 + rr1 + hi * 8;
                    const int col  = colBase0 + sn * 16 + lm;
                    const bool offd = (row != col);
                    sum += offd ? s : 0.0f;
                    mx   = fmaxf(mx, offd ? s : -4.0f);
                    cnt += (unsigned int)(offd & hot);
                    bal[sn] = __builtin_amdgcn_ballot_w32(hot | (!offd));
                } else {
                    sum += s;
                    mx   = fmaxf(mx, s);
                    cnt += (unsigned int)hot;
                    bal[sn] = __builtin_amdgcn_ballot_w32(hot);
                }
            }
            /* compose the 4 uniform 32-bit adjacency words for rows rr1/rr2 */
            const unsigned int w0r1 = (bal[0] & 0xffffu) | (bal[1] << 16);
            const unsigned int w1r1 = (bal[2] & 0xffffu) | (bal[3] << 16);
            const unsigned int w0r2 = (bal[0] >> 16) | (bal[1] & 0xffff0000u);
            const unsigned int w1r2 = (bal[2] >> 16) | (bal[3] & 0xffff0000u);
            /* lane-deposit: v_cmp + v_cndmask (uniform SGPR sources), no branches */
            const bool is1 = (lane == rr1);
            const bool is2 = (lane == rr2);
            wlo = is1 ? w0r1 : (is2 ? w0r2 : wlo);
            whi = is1 ? w1r1 : (is2 ? w1r2 : whi);
            SCHED_FENCE();   /* consume this group's masks before the next */
        }
    }
}

/* ---------------- WMMA sim tiles -> stats + adjacency bits ---------------- */
__global__ __launch_bounds__(256) void k_simtiles(const _Float16* __restrict__ normed,
                                                  unsigned int* __restrict__ adj,
                                                  unsigned int* __restrict__ su32,
                                                  unsigned long long* __restrict__ su64) {
    const int lane  = threadIdx.x & 31;
    const int wave  = threadIdx.x >> 5;
    const int waveM = wave & 3;                 /* 4 waves along M: 32 rows each */
    const int waveN = wave >> 2;                /* 2 waves along N: 64 cols each */
    const int tileM = blockIdx.y * 128;
    const int tileN = blockIdx.x * 128;
    const int rowBase0 = tileM + waveM * 32;
    const int colBase0 = tileN + waveN * 64;
    const int lm = lane & 15;
    const int hi = lane >> 4;                   /* half-wave select: 0 or 1 */

    v8f acc[2][4];
    const v8f vzero = {0.f, 0.f, 0.f, 0.f, 0.f, 0.f, 0.f, 0.f};
#pragma unroll
    for (int a = 0; a < 2; a++)
#pragma unroll
        for (int b = 0; b < 4; b++) acc[a][b] = vzero;

    for (int kk = 0; kk < DDIM; kk += 32) {
        /* A fragment, 16x32 f16 layout: lanes<16 hold K kk..kk+7 & kk+16..kk+23,
           lanes>=16 hold K kk+8..kk+15 & kk+24..kk+31 */
        v16h afrag[2];
#pragma unroll
        for (int sm = 0; sm < 2; sm++) {
            int row = rowBase0 + sm * 16 + lm;
            const _Float16* p = normed + row * DDIM + kk + hi * 8;
            v8h lo = *(const v8h*)p;
            v8h hb = *(const v8h*)(p + 16);
            v16h a;
#pragma unroll
            for (int i = 0; i < 8; i++) { a[i] = lo[i]; a[i + 8] = hb[i]; }
            afrag[sm] = a;
        }
        /* B fragment, 32x16 f16: lane n holds column n (== row n of normed, since B=normed^T),
           lanes<16: K kk..kk+7 & kk+16..kk+23, lanes>=16: K kk+8..kk+15 & kk+24..kk+31 */
        v16h bfrag[4];
#pragma unroll
        for (int sn = 0; sn < 4; sn++) {
            int col = colBase0 + sn * 16 + lm;
            const _Float16* q = normed + col * DDIM + kk + hi * 8;
            v8h blo = *(const v8h*)q;
            v8h bhi = *(const v8h*)(q + 16);
            v16h b;
#pragma unroll
            for (int i = 0; i < 8; i++) { b[i] = blo[i]; b[i + 8] = bhi[i]; }
            bfrag[sn] = b;
        }
#pragma unroll
        for (int sm = 0; sm < 2; sm++)
#pragma unroll
            for (int sn = 0; sn < 4; sn++)
                acc[sm][sn] = __builtin_amdgcn_wmma_f32_16x16x32_f16(
                    false, afrag[sm], false, bfrag[sn], (short)0, acc[sm][sn], false, false);
    }

    float mx = -4.0f, sum = 0.0f;
    unsigned int cnt = 0;
    unsigned int wlo = 0, whi = 0;
    const int wb = colBase0 >> 5;

    /* single uniform branch; each path is fully straight-line */
    if (blockIdx.x == blockIdx.y)
        tile_epilogue<true >(acc, lane, lm, hi, rowBase0, colBase0, mx, sum, cnt, wlo, whi);
    else
        tile_epilogue<false>(acc, lane, lm, hi, rowBase0, colBase0, mx, sum, cnt, wlo, whi);

    /* lane L -> row rowBase0+L, words [wb, wb+1]: one b64 store, no divergence */
    {
        uint2 v; v.x = wlo; v.y = whi;
        *(uint2*)(adj + (size_t)(rowBase0 + lane) * WPR + wb) = v;
    }

    /* wave-level shuffle reduction, then 8 partials through LDS */
#pragma unroll
    for (int off = 16; off > 0; off >>= 1) {
        mx  = fmaxf(mx, __shfl_xor(mx, off, 32));
        sum += __shfl_xor(sum, off, 32);
        cnt += __shfl_xor(cnt, off, 32);
    }
    __shared__ float smax[8], ssum[8];
    __shared__ unsigned int scnt[8];
    if (lane == 0) { smax[wave] = mx; ssum[wave] = sum; scnt[wave] = cnt; }
    __syncthreads();
    if (threadIdx.x == 0) {
        float m2 = smax[0], s2 = ssum[0];
        unsigned int c2 = scnt[0];
#pragma unroll
        for (int i = 1; i < 8; i++) {
            m2 = fmaxf(m2, smax[i]); s2 += ssum[i]; c2 += scnt[i];
        }
        atomicMax(&su32[0], fkey(m2));
        atomicAdd(&su32[1], c2);
        long long fx = (long long)((double)s2 * 1048576.0);   /* fixed-point 2^20 */
        atomicAdd(&su64[1], (unsigned long long)fx);
    }
}

/* ---------------- label propagation: one wave per row ---------------- */
__global__ __launch_bounds__(256) void k_minprop(const unsigned int* __restrict__ adj,
                                                 const int* __restrict__ lin,
                                                 int* __restrict__ lout) {
    int lane = threadIdx.x & 31;
    int row  = blockIdx.x * 8 + (threadIdx.x >> 5);
    int m = lin[row];
#pragma unroll
    for (int w = lane; w < WPR; w += 32) {
        unsigned int bits = adj[row * WPR + w];
        while (bits) {
            int b = __builtin_ctz(bits);
            bits &= bits - 1;
            m = min(m, lin[w * 32 + b]);
        }
    }
#pragma unroll
    for (int off = 16; off > 0; off >>= 1)
        m = min(m, __shfl_xor(m, off, 32));
    if (lane == 0) lout[row] = m;
}

__global__ void k_jump(const int* __restrict__ lin, int* __restrict__ lout) {
    int i = blockIdx.x * blockDim.x + threadIdx.x;
    if (i < NROWS) lout[i] = lin[lin[i]];
}

/* ---------------- features + tiny MLP head ---------------- */
__global__ __launch_bounds__(256) void k_final(const int* __restrict__ lab,
                                               const unsigned int* __restrict__ su32,
                                               const unsigned long long* __restrict__ su64,
                                               const float* __restrict__ w1,
                                               const float* __restrict__ b1,
                                               const float* __restrict__ w2,
                                               const float* __restrict__ b2,
                                               float* __restrict__ out) {
    __shared__ unsigned int sred[256];
    int t = threadIdx.x;
    unsigned int roots = 0;
    for (int i = t; i < NROWS; i += 256) roots += (lab[i] == i) ? 1u : 0u;
    sred[t] = roots;
    __syncthreads();
    for (int s = 128; s > 0; s >>= 1) {
        if (t < s) sred[t] += sred[t + s];
        __syncthreads();
    }
    if (t == 0) {
        const double n_pairs = (double)NROWS * (double)(NROWS - 1);
        float max_sim  = fdec(su32[0]);
        double sumd    = (double)(long long)su64[1] / 1048576.0;
        float mean_sim = (float)(sumd / n_pairs);
        float frac     = (float)((double)su32[1] / n_pairs);
        float ccount   = (float)sred[0] / (float)NROWS;
        float f[4] = {max_sim, mean_sim, frac, ccount};
        float h[16];
        for (int j = 0; j < 16; j++) {
            float z = b1[j];
            for (int k = 0; k < 4; k++) z += f[k] * w1[k * 16 + j];
            h[j] = 0.5f * z * (1.0f + erff(z * 0.7071067811865475f));  /* exact gelu */
        }
        float a = b2[0];
        for (int j = 0; j < 16; j++) a += h[j] * w2[j];
        out[0] = 1.0f / (1.0f + expf(-a));
    }
}

extern "C" void kernel_launch(void* const* d_in, const int* in_sizes, int n_in,
                              void* d_out, int out_size, void* d_ws, size_t ws_size,
                              hipStream_t stream) {
    const float* emb = (const float*)d_in[0];
    const float* w1  = (const float*)d_in[1];
    const float* b1  = (const float*)d_in[2];
    const float* w2  = (const float*)d_in[3];
    const float* b2  = (const float*)d_in[4];

    char* ws = (char*)d_ws;
    _Float16* normed          = (_Float16*)(ws + OFF_NORMED);
    unsigned int* adj         = (unsigned int*)(ws + OFF_ADJ);
    int* lab0                 = (int*)(ws + OFF_LAB0);
    int* lab1                 = (int*)(ws + OFF_LAB1);
    unsigned int* su32        = (unsigned int*)(ws + OFF_STATS);
    unsigned long long* su64  = (unsigned long long*)(ws + OFF_STATS);

    k_init<<<NROWS / 256, 256, 0, stream>>>(lab0, su32, su64);
    k_normalize<<<NROWS, 256, 0, stream>>>(emb, normed);

    dim3 grid(64, 64);  /* 128x128 tiles over 8192x8192 */
    k_simtiles<<<grid, 256, 0, stream>>>(normed, adj, su32, su64);

    /* iters = ceil(log2(8192)) + 3 = 16 */
    for (int it = 0; it < 16; ++it) {
        k_minprop<<<NROWS / 8, 256, 0, stream>>>(adj, lab0, lab1);
        k_jump<<<NROWS / 256, 256, 0, stream>>>(lab1, lab0);
    }
    k_final<<<1, 256, 0, stream>>>(lab0, su32, su64, w1, b1, w2, b2, (float*)d_out);
}
